// TriadicClosureModule_6305011991196
// MI455X (gfx1250) — matrix operand order repaired and checked by reading, not compiled, not run on hardware
//
#include <hip/hip_runtime.h>
#include <hip/hip_bf16.h>

typedef __attribute__((ext_vector_type(16))) _Float16 v16h;
typedef __attribute__((ext_vector_type(8)))  _Float16 v8h;
typedef __attribute__((ext_vector_type(8)))  float    v8f;
typedef __attribute__((ext_vector_type(4)))  unsigned int u32x4;
typedef __attribute__((ext_vector_type(4)))  int      v4i;

#define N_NODES 8192
#define N_PAIRS 16384
#define DIM     256
#define JBLK    4     // fallback kernel: j-tiles per wave (16x64 strip)
#define ROWH    40    // fast kernel: padded LDS row stride in halves (32 data + 8 pad = 80B)

// ---- CDNA5 async global->LDS path (guarded; falls back to load + ds_store) ----
#if defined(__has_builtin)
#  if __has_builtin(__builtin_amdgcn_global_load_async_to_lds_b128)
#    define HAVE_ASYNC_LDS 1
#  endif
#endif

#ifdef HAVE_ASYNC_LDS
#  if __has_builtin(__builtin_amdgcn_s_wait_asynccnt)
#    define WAIT_ASYNC0() __builtin_amdgcn_s_wait_asynccnt(0)
#  else
#    define WAIT_ASYNC0() asm volatile("s_wait_asynccnt 0" ::: "memory")
#  endif
// Builtin signature (from hipcc diagnostic): param 0 is `v4i __device__*` (AS1).
typedef v4i __attribute__((address_space(1)))* gv4i_p;   // pointer to int4 in global AS
typedef v4i __attribute__((address_space(3)))* lv4i_p;   // pointer to int4 in LDS AS
#  define GLB_CAST(p) ((gv4i_p)(size_t)(p))
#  define LDS_CAST(p) ((lv4i_p)(unsigned int)(size_t)(p))
#endif

// ---------------------------------------------------------------- zero init
__global__ void zero_f32(float* __restrict__ p, int n) {
  int i = blockIdx.x * blockDim.x + threadIdx.x;
  if (i < n) p[i] = 0.f;
}

// ---------------------------------------------------------------- adj (f32) -> adjh (f16), exact for {0,1}
__global__ __launch_bounds__(256) void convert_f16_kernel(const float* __restrict__ a,
                                                          _Float16* __restrict__ h) {
  size_t i = ((size_t)blockIdx.x * 256 + threadIdx.x) * 4;
  const float* p = a + i;
  _Float16* q = h + i;
  q[0] = (_Float16)p[0];
  q[1] = (_Float16)p[1];
  q[2] = (_Float16)p[2];
  q[3] = (_Float16)p[3];
}

// ---------------------------------------------------------------- degree[i] = sum_j adj[i,j]
__global__ __launch_bounds__(256) void degree_kernel(const float* __restrict__ adj,
                                                     float* __restrict__ deg) {
  const int i = blockIdx.x;
  const float* row = adj + (size_t)i * N_NODES;
  float s = 0.f;
  for (int j = threadIdx.x; j < N_NODES; j += 256) s += row[j];
  __shared__ float sm[256];
  sm[threadIdx.x] = s;
  __syncthreads();
  for (int st = 128; st > 0; st >>= 1) {
    if (threadIdx.x < st) sm[threadIdx.x] += sm[threadIdx.x + st];
    __syncthreads();
  }
  if (threadIdx.x == 0) deg[i] = sm[0];
}

// ================================================================ FAST PATH
// tri[i] = sum_j (A@A)[i,j] * A[j,i], A pre-converted to f16 (exact {0,1}).
// Workgroup = 4 waves; wave w owns i-tile i0=bi*64+16w; all waves share the
// 64x32 f16 B tile (rows j, using adjacency symmetry) staged in LDS,
// double-buffered via GLOBAL_LOAD_ASYNC_TO_LDS_B128 when available.
__global__ __launch_bounds__(128) void tri_wmma_lds_kernel(const _Float16* __restrict__ adjh,
                                                           float* __restrict__ tri) {
  __shared__ __align__(16) _Float16 ldsB[2][64 * ROWH];

  const int bj = blockIdx.x & 127;
  const int bi = blockIdx.x >> 7;
  const int j0 = bj * 64;
  const int wave = threadIdx.x >> 5;
  const int lane = threadIdx.x & 31;
  const int i0 = bi * 64 + wave * 16;
  const int rm = lane & 15;
  const int hi = lane >> 4;

  // Cooperative staging map: 64 rows x 64B (4 chunks of 16B) = 256 chunks;
  // thread t moves chunks t and t+128.
  const int tid = threadIdx.x;
  const int r0 = tid >> 2, q0 = tid & 3;
  const int r1 = (tid + 128) >> 2;
  const size_t g0 = (size_t)(j0 + r0) * N_NODES + q0 * 8;
  const size_t g1 = (size_t)(j0 + r1) * N_NODES + q0 * 8;
  const int l0 = r0 * ROWH + q0 * 8;
  const int l1 = r1 * ROWH + q0 * 8;

  // A operand (16x32 MxK): row i0+rm; halves 0..15 = K (hi?16:0)..+15
  const _Float16* arow = adjh + (size_t)(i0 + rm) * N_NODES + (hi ? 16 : 0);

  const v8f vz = {};
  v8f acc0 = vz, acc1 = vz, acc2 = vz, acc3 = vz;
  int cur = 0;

#ifdef HAVE_ASYNC_LDS
  __builtin_amdgcn_global_load_async_to_lds_b128(GLB_CAST(adjh + g0), LDS_CAST(&ldsB[0][l0]), 0, 0);
  __builtin_amdgcn_global_load_async_to_lds_b128(GLB_CAST(adjh + g1), LDS_CAST(&ldsB[0][l1]), 0, 0);
  WAIT_ASYNC0();
  __syncthreads();
#endif

  for (int k0 = 0; k0 < N_NODES; k0 += 32) {
#ifdef HAVE_ASYNC_LDS
    const int nxt = cur ^ 1;
    if (k0 + 32 < N_NODES) {
      __builtin_amdgcn_global_load_async_to_lds_b128(GLB_CAST(adjh + g0 + k0 + 32),
                                                     LDS_CAST(&ldsB[nxt][l0]), 0, 0);
      __builtin_amdgcn_global_load_async_to_lds_b128(GLB_CAST(adjh + g1 + k0 + 32),
                                                     LDS_CAST(&ldsB[nxt][l1]), 0, 0);
    }
#else
    // Fallback staging: plain global load -> ds_store, two barriers.
    u32x4 s0 = *(const u32x4*)(adjh + g0 + k0);
    u32x4 s1 = *(const u32x4*)(adjh + g1 + k0);
    *(u32x4*)&ldsB[0][l0] = s0;
    *(u32x4*)&ldsB[0][l1] = s1;
    __syncthreads();
#endif
    // A fragment from global (L2-resident f16)
    v16h a;
    *(v8h*)&a         = *(const v8h*)(arow + k0);
    *(((v8h*)&a) + 1) = *(const v8h*)(arow + k0 + 8);

    // B fragments from LDS: lane = column n=rm of tile jt; halves 0..15 = K (hi?16:0)..+15
    const _Float16* lb = &ldsB[cur][rm * ROWH + hi * 16];
    v16h b0, b1, b2, b3;
    *(v8h*)&b0         = *(const v8h*)(lb);
    *(((v8h*)&b0) + 1) = *(const v8h*)(lb + 8);
    *(v8h*)&b1         = *(const v8h*)(lb + 16 * ROWH);
    *(((v8h*)&b1) + 1) = *(const v8h*)(lb + 16 * ROWH + 8);
    *(v8h*)&b2         = *(const v8h*)(lb + 32 * ROWH);
    *(((v8h*)&b2) + 1) = *(const v8h*)(lb + 32 * ROWH + 8);
    *(v8h*)&b3         = *(const v8h*)(lb + 48 * ROWH);
    *(((v8h*)&b3) + 1) = *(const v8h*)(lb + 48 * ROWH + 8);

    // Back-to-back WMMAs: shared A (read-read), disjoint D -> no RAW hazard nops
    acc0 = __builtin_amdgcn_wmma_f32_16x16x32_f16(false, a, false, b0, (short)0, acc0, false, false);
    acc1 = __builtin_amdgcn_wmma_f32_16x16x32_f16(false, a, false, b1, (short)0, acc1, false, false);
    acc2 = __builtin_amdgcn_wmma_f32_16x16x32_f16(false, a, false, b2, (short)0, acc2, false, false);
    acc3 = __builtin_amdgcn_wmma_f32_16x16x32_f16(false, a, false, b3, (short)0, acc3, false, false);

#ifdef HAVE_ASYNC_LDS
    WAIT_ASYNC0();
    __syncthreads();
    cur = nxt;
#else
    __syncthreads();
#endif
  }

  // D layout: lane, elem v -> m = v + 8*hi, n = rm.
  // tri[i0+m] += sum_n D[m,n] * A[j,i]  (A symmetric -> contiguous f16 read)
#pragma unroll
  for (int v = 0; v < 8; ++v) {
    const int m = v + 8 * hi;
    const _Float16* maskrow = adjh + (size_t)(i0 + m) * N_NODES + j0;
    float val = acc0[v] * (float)maskrow[rm]      + acc1[v] * (float)maskrow[16 + rm] +
                acc2[v] * (float)maskrow[32 + rm] + acc3[v] * (float)maskrow[48 + rm];
    val += __shfl_xor(val, 1, 32);
    val += __shfl_xor(val, 2, 32);
    val += __shfl_xor(val, 4, 32);
    val += __shfl_xor(val, 8, 32);
    if (rm == 0) atomicAdd(&tri[i0 + m], val);
  }
}

// ================================================================ FALLBACK PATH
// (known-good round-1 kernel: in-register f32->f16 conversion, 16x64 strip/wave)
__global__ __launch_bounds__(32) void tri_wmma_kernel(const float* __restrict__ adj,
                                                      float* __restrict__ tri) {
  const int jChunks = N_NODES / (16 * JBLK);
  const int i0 = (int)(blockIdx.x / jChunks) * 16;
  const int j0 = (int)(blockIdx.x % jChunks) * (16 * JBLK);
  const int lane = threadIdx.x;
  const int rm   = lane & 15;
  const int hi   = lane >> 4;

  const float* arow  = adj + (size_t)(i0 + rm) * N_NODES + (hi ? 16 : 0);
  const float* brow0 = adj + (size_t)(j0 + rm) * N_NODES + (hi ? 16 : 0);

  const v8f vzero = {};
  v8f acc[JBLK];
#pragma unroll
  for (int jt = 0; jt < JBLK; ++jt) acc[jt] = vzero;

  for (int k0 = 0; k0 < N_NODES; k0 += 32) {
    v16h a;
#pragma unroll
    for (int t = 0; t < 8; ++t) {
      a[t]     = (_Float16)arow[k0 + t];
      a[t + 8] = (_Float16)arow[k0 + 8 + t];
    }
#pragma unroll
    for (int jt = 0; jt < JBLK; ++jt) {
      const float* brow = brow0 + (size_t)(16 * jt) * N_NODES;
      v16h b;
#pragma unroll
      for (int t = 0; t < 16; ++t) b[t] = (_Float16)brow[k0 + t];
      acc[jt] = __builtin_amdgcn_wmma_f32_16x16x32_f16(
          false, a, false, b, (short)0, acc[jt], false, false);
    }
  }

#pragma unroll
  for (int v = 0; v < 8; ++v) {
    const int m = v + 8 * hi;
    const float* maskrow = adj + (size_t)(i0 + m) * N_NODES + j0;
    float val = 0.f;
#pragma unroll
    for (int jt = 0; jt < JBLK; ++jt) val += acc[jt][v] * maskrow[16 * jt + rm];
    val += __shfl_xor(val, 1, 32);
    val += __shfl_xor(val, 2, 32);
    val += __shfl_xor(val, 4, 32);
    val += __shfl_xor(val, 8, 32);
    if (rm == 0) atomicAdd(&tri[i0 + m], val);
  }
}

// ---------------------------------------------------------------- w[i] = C[i] * g[i]
__global__ __launch_bounds__(256) void w_kernel(const float* __restrict__ embeds,
                                                const float* __restrict__ deg,
                                                const float* __restrict__ tri,
                                                float* __restrict__ w) {
  const int i = blockIdx.x;
  float v = embeds[(size_t)i * DIM + threadIdx.x];
  float diff = v - v;
  float s = diff * diff;
  __shared__ float sm[256];
  sm[threadIdx.x] = s;
  __syncthreads();
  for (int st = 128; st > 0; st >>= 1) {
    if (threadIdx.x < st) sm[threadIdx.x] += sm[threadIdx.x + st];
    __syncthreads();
  }
  if (threadIdx.x == 0) {
    float g = -sm[0];
    float d = deg[i];
    float den = d * (d - 1.f);
    if (den == 0.f) den = 1e-6f;
    float C = 2.f * tri[i] / den;
    w[i] = C * g;
  }
}

// ---------------------------------------------------------------- lambda[p] = sum_n adj[m,n]*adj[q,n]*w[n]
__global__ __launch_bounds__(256) void lambda_kernel(const int* __restrict__ pairs,
                                                     const float* __restrict__ adj,
                                                     const float* __restrict__ w,
                                                     float* __restrict__ out) {
  const int p = blockIdx.x;
  const int m = pairs[2 * p];
  const int q = pairs[2 * p + 1];
  const float* rowm = adj + (size_t)m * N_NODES;
  const float* rowq = adj + (size_t)q * N_NODES;
  float s = 0.f;
  for (int n = threadIdx.x; n < N_NODES; n += 256)
    s += rowm[n] * rowq[n] * w[n];
  __shared__ float sm[256];
  sm[threadIdx.x] = s;
  __syncthreads();
  for (int st = 128; st > 0; st >>= 1) {
    if (threadIdx.x < st) sm[threadIdx.x] += sm[threadIdx.x + st];
    __syncthreads();
  }
  if (threadIdx.x == 0) out[p] = sm[0];
}

// ---------------------------------------------------------------- launch
extern "C" void kernel_launch(void* const* d_in, const int* in_sizes, int n_in,
                              void* d_out, int out_size, void* d_ws, size_t ws_size,
                              hipStream_t stream) {
  (void)in_sizes; (void)n_in; (void)out_size;

  const int*   pairs  = (const int*)d_in[0];    // node_pairs (P,2) int32
  const float* adj    = (const float*)d_in[1];  // adj_matrix (N,N) f32
  const float* embeds = (const float*)d_in[2];  // node_embeds (N,D) f32
  // d_in[3] = event_history, unused by the reference computation

  float* tri = (float*)d_ws;        // N floats
  float* deg = tri + N_NODES;       // N floats
  float* w   = deg + N_NODES;       // N floats
  const size_t headBytes = 3 * (size_t)N_NODES * sizeof(float);    // 96 KB, 256-aligned
  _Float16* adjh = (_Float16*)((char*)d_ws + headBytes);
  const size_t needFast = headBytes + (size_t)N_NODES * N_NODES * sizeof(_Float16);
  const bool fast = (ws_size >= needFast);

  float* out = (float*)d_out;

  zero_f32<<<(N_NODES + 255) / 256, 256, 0, stream>>>(tri, N_NODES);
  degree_kernel<<<N_NODES, 256, 0, stream>>>(adj, deg);

  if (fast) {
    const size_t total = (size_t)N_NODES * N_NODES;        // 64Mi elements
    convert_f16_kernel<<<(unsigned)(total / (256 * 4)), 256, 0, stream>>>(adj, adjh);
    const int blocks = (N_NODES / 64) * (N_NODES / 64);    // 128*128 = 16384
    tri_wmma_lds_kernel<<<blocks, 128, 0, stream>>>(adjh, tri);
  } else {
    const int jChunks = N_NODES / (16 * JBLK);             // 128
    const int iTiles  = N_NODES / 16;                      // 512
    tri_wmma_kernel<<<iTiles * jChunks, 32, 0, stream>>>(adj, tri);
  }

  w_kernel<<<N_NODES, 256, 0, stream>>>(embeds, deg, tri, w);
  lambda_kernel<<<N_PAIRS, 256, 0, stream>>>(pairs, adj, w, out);
}
